// AttentionHead_40922448396312
// MI455X (gfx1250) — matrix operand order repaired and checked by reading, not compiled, not run on hardware
//
#include <hip/hip_runtime.h>

#define N_TOK 8192
#define D_IN  1024
#define D_H   64

typedef __attribute__((ext_vector_type(16))) __bf16       v16bf;
typedef __attribute__((ext_vector_type(8)))  float        v8f;
typedef __attribute__((ext_vector_type(8)))  unsigned int v8u;
typedef __attribute__((ext_vector_type(4)))  unsigned int v4u;

#if defined(__has_builtin)
#  if __has_builtin(__builtin_amdgcn_cvt_pk_bf16_f32)
#    define HAVE_CVT_PK_BF16 1
#  else
#    define HAVE_CVT_PK_BF16 0
#  endif
#else
#  define HAVE_CVT_PK_BF16 0
#endif

static __device__ __forceinline__ unsigned short bf16_manual(float f) {
  unsigned int u = __builtin_bit_cast(unsigned int, f);
  u = (u + 0x7FFFu + ((u >> 16) & 1u)) >> 16;   // round-to-nearest-even
  return (unsigned short)u;
}
static __device__ __forceinline__ unsigned int bf16_pk(float lo, float hi) {
#if HAVE_CVT_PK_BF16
  auto v = __builtin_amdgcn_cvt_pk_bf16_f32(lo, hi);   // v_cvt_pk_bf16_f32
  return __builtin_bit_cast(unsigned int, v);
#else
  return (unsigned int)bf16_manual(lo) | ((unsigned int)bf16_manual(hi) << 16);
#endif
}
static __device__ __forceinline__ unsigned short bf16_1(float f) {
#if HAVE_CVT_PK_BF16
  return (unsigned short)bf16_pk(f, 0.0f);
#else
  return bf16_manual(f);
#endif
}

// Issue a TDM tile DMA: D# group0 (4 SGPRs) + group1 (8 SGPRs), groups 2/3 NULL.
#define TDM_ISSUE(g0v, g1v) \
  asm volatile("tensor_load_to_lds %0, %1" :: "s"(g0v), "s"(g1v) : "memory")

// ---------------------------------------------------------------------------
// Kernel 0a: x (fp32, row-major) -> xb (bf16, row-major). Pure bandwidth.
// ---------------------------------------------------------------------------
__global__ __launch_bounds__(256) void cvt_x_kernel(
    const float* __restrict__ x, unsigned short* __restrict__ xb)
{
  const size_t i = ((size_t)blockIdx.x * 256 + threadIdx.x) * 4;
  const float4 f = *(const float4*)(x + i);
  uint2 r;
  r.x = bf16_pk(f.x, f.y);
  r.y = bf16_pk(f.z, f.w);
  *(uint2*)(xb + i) = r;
}

// ---------------------------------------------------------------------------
// Kernel 0b: W[1024,64] fp32 -> WT[64,1024] bf16 (transposed) for q,k,v.
// ---------------------------------------------------------------------------
__global__ __launch_bounds__(256) void cvt_w_kernel(
    const float* __restrict__ Wq, const float* __restrict__ Wk,
    const float* __restrict__ Wv,
    unsigned short* __restrict__ wtq, unsigned short* __restrict__ wtk,
    unsigned short* __restrict__ wtv)
{
  const int which = blockIdx.x >> 8;                        // 0,1,2
  const int idx   = (blockIdx.x & 255) * 256 + threadIdx.x; // 0..65535
  const int k = idx >> 6;                                   // 0..1023
  const int c = idx & 63;                                   // 0..63
  const float*    W  = (which == 0) ? Wq  : (which == 1) ? Wk  : Wv;
  unsigned short* WT = (which == 0) ? wtq : (which == 1) ? wtk : wtv;
  WT[(size_t)c * D_IN + k] = bf16_1(W[(size_t)k * D_H + c]);
}

// ---------------------------------------------------------------------------
// Kernel 1: q/k/v projection GEMMs, all three on v_wmma_f32_16x16x32_bf16.
// ---------------------------------------------------------------------------
__global__ __launch_bounds__(128) void qkv_kernel(
    const unsigned short* __restrict__ xb,
    const unsigned short* __restrict__ wtq,
    const unsigned short* __restrict__ wtk,
    const unsigned short* __restrict__ wtv,
    const float* __restrict__ bq, const float* __restrict__ bk,
    const float* __restrict__ bv,
    unsigned short* __restrict__ qo,
    unsigned short* __restrict__ ko,
    unsigned short* __restrict__ vT)
{
  const int lane = threadIdx.x & 31;
  const int wave = threadIdx.x >> 5;
  const int half = lane >> 4;
  const int l16  = lane & 15;
  const int m0   = blockIdx.x * 16;
  const int n0   = wave * 16;

  v8f accq = {}; v8f acck = {}; v8f accv = {};
  const unsigned short* xrow = xb  + (size_t)(m0 + l16) * D_IN;
  const unsigned short* wqr  = wtq + (size_t)(n0 + l16) * D_IN;
  const unsigned short* wkr  = wtk + (size_t)(n0 + l16) * D_IN;
  const unsigned short* wvr  = wtv + (size_t)(n0 + l16) * D_IN;

  for (int kk = 0; kk < D_IN; kk += 32) {
    const uint4 a0 = *(const uint4*)(xrow + kk + half * 16);
    const uint4 a1 = *(const uint4*)(xrow + kk + half * 16 + 8);
    v8u au = {a0.x, a0.y, a0.z, a0.w, a1.x, a1.y, a1.z, a1.w};
    const v16bf A = __builtin_bit_cast(v16bf, au);

#define LOAD_B(ptr, dst)                                              \
    {                                                                 \
      const uint4 b0_ = *(const uint4*)((ptr) + kk + half * 16);      \
      const uint4 b1_ = *(const uint4*)((ptr) + kk + half * 16 + 8);  \
      v8u bu_ = {b0_.x, b0_.y, b0_.z, b0_.w, b1_.x, b1_.y, b1_.z, b1_.w}; \
      dst = __builtin_bit_cast(v16bf, bu_);                           \
    }
    v16bf Bq, Bk, Bv;
    LOAD_B(wqr, Bq)
    LOAD_B(wkr, Bk)
    LOAD_B(wvr, Bv)
#undef LOAD_B

    accq = __builtin_amdgcn_wmma_f32_16x16x32_bf16(false, A, false, Bq,
                                                   (short)0, accq, false, false);
    acck = __builtin_amdgcn_wmma_f32_16x16x32_bf16(false, A, false, Bk,
                                                   (short)0, acck, false, false);
    accv = __builtin_amdgcn_wmma_f32_16x16x32_bf16(false, A, false, Bv,
                                                   (short)0, accv, false, false);
  }

  const float SCALE = 0.125f;                 // 1/sqrt(64), folded into q
  const float biasq = bq[n0 + l16];
  const float biask = bk[n0 + l16];
  const float biasv = bv[n0 + l16];
#pragma unroll
  for (int r = 0; r < 8; ++r) {               // C layout: M = r + half*8, N = l16
    const int row = m0 + r + half * 8;
    const int col = n0 + l16;
    qo[(size_t)row * D_H + col]   = bf16_1((accq[r] + biasq) * SCALE);
    ko[(size_t)row * D_H + col]   = bf16_1(acck[r] + biask);
    vT[(size_t)col * N_TOK + row] = bf16_1(accv[r] + biasv);
  }
}

// ---------------------------------------------------------------------------
// Kernel 2: flash attention. K and V tiles are streamed into LDS by the
// Tensor Data Mover with a ping-pong double buffer (TENSORcnt is in-order
// per wave, so waiting to <=2 guarantees the current step's two DMAs landed
// while the next step's are still in flight).
// ---------------------------------------------------------------------------
__global__ __launch_bounds__(64) void attn_kernel(
    const unsigned short* __restrict__ qo,
    const unsigned short* __restrict__ ko,
    const unsigned short* __restrict__ vT,
    float* __restrict__ out)
{
  __shared__ __align__(16) unsigned short lds_k[2][2][32][64]; // [wave][buf] K tiles
  __shared__ __align__(16) unsigned short lds_v[2][2][64][32]; // [wave][buf] V^T tiles
  __shared__ __align__(16) unsigned short lds_p[2][16][32];    // [wave] P tiles

  const int lane = threadIdx.x & 31;
  const int wave = threadIdx.x >> 5;
  const int half = lane >> 4;
  const int l16  = lane & 15;
  const int m0   = (blockIdx.x * 2 + wave) * 16;

  // Q fragments resident in registers for the whole key loop
  v16bf qa[2];
  {
    const unsigned short* qrow = qo + (size_t)(m0 + l16) * D_H;
#pragma unroll
    for (int s = 0; s < 2; ++s) {
      const uint4 u0 = *(const uint4*)(qrow + s * 32 + half * 16);
      const uint4 u1 = *(const uint4*)(qrow + s * 32 + half * 16 + 8);
      v8u a = {u0.x, u0.y, u0.z, u0.w, u1.x, u1.y, u1.z, u1.w};
      qa[s] = __builtin_bit_cast(v16bf, a);
    }
  }

  v8f o[4] = {v8f{}, v8f{}, v8f{}, v8f{}};
  float m[8], l[8];
#pragma unroll
  for (int r = 0; r < 8; ++r) { m[r] = -3.0e38f; l[r] = 0.0f; }

  // --- loop-invariant D# group1 descriptors (ISA 8.4) ---
  // K: tensor ko[8192][64] bf16 (row stride 64); tile 32 rows x 64 cols.
  const v8u g1_k = {0x00010000u,                    // data_size=1 (2 bytes)
                    (unsigned)D_H << 16,            // tensor_dim0 = 64
                    (unsigned)(N_TOK & 0xFFFF) << 16, // tensor_dim1 = 8192
                    (unsigned)D_H << 16,            // tile_dim0 = 64
                    32u,                            // tile_dim1 = 32
                    (unsigned)D_H,                  // tensor_dim0_stride = 64
                    0u, 0u};
  // V: tensor vT[64][8192] bf16 (row stride 8192); tile 64 rows x 32 cols.
  const v8u g1_v = {0x00010000u,                    // data_size=1 (2 bytes)
                    (unsigned)(N_TOK & 0xFFFF) << 16, // tensor_dim0 = 8192
                    (unsigned)D_H << 16,            // tensor_dim1 = 64
                    32u << 16,                      // tile_dim0 = 32
                    (unsigned)D_H,                  // tile_dim1 = 64
                    (unsigned)N_TOK,                // tensor_dim0_stride = 8192
                    0u, 0u};
  const unsigned long long ko_base = (unsigned long long)(uintptr_t)ko;
  const unsigned long long vT_base = (unsigned long long)(uintptr_t)vT;
  const unsigned koff[2] = {(unsigned)(size_t)(void*)&lds_k[wave][0][0][0],
                            (unsigned)(size_t)(void*)&lds_k[wave][1][0][0]};
  const unsigned voff[2] = {(unsigned)(size_t)(void*)&lds_v[wave][0][0][0],
                            (unsigned)(size_t)(void*)&lds_v[wave][1][0][0]};

#define ISSUE_STEP(step, buf)                                                 \
  {                                                                           \
    const unsigned long long gaK = ko_base + (unsigned long long)(step) * 32u * (D_H * 2); \
    const v4u g0k = {1u, koff[buf], (unsigned)(gaK & 0xFFFFFFFFu),            \
                     ((unsigned)(gaK >> 32) & 0x01FFFFFFu) | 0x80000000u};    \
    TDM_ISSUE(g0k, g1_k);                                                     \
    const unsigned long long gaV = vT_base + (unsigned long long)(step) * 32u * 2u; \
    const v4u g0v = {1u, voff[buf], (unsigned)(gaV & 0xFFFFFFFFu),            \
                     ((unsigned)(gaV >> 32) & 0x01FFFFFFu) | 0x80000000u};    \
    TDM_ISSUE(g0v, g1_v);                                                     \
  }

  const int NSTEP = N_TOK / 32;
  ISSUE_STEP(0, 0)                               // prologue: DMA step 0

  for (int it = 0; it < NSTEP; ++it) {
    const int buf = it & 1;
    const int kt  = it * 32;

    if (it + 1 < NSTEP) {
      // Reads of buf^1 (from step it-1) must be done before we overwrite it.
      asm volatile("s_wait_dscnt 0" ::: "memory");
      ISSUE_STEP(it + 1, buf ^ 1)
      asm volatile("s_wait_tensorcnt 2" ::: "memory"); // step `it` tiles landed
    } else {
      asm volatile("s_wait_tensorcnt 0" ::: "memory");
    }

    // ---- S = Q @ K^T, K fragments from the LDS tile (ds_load_b128) ----
    v8f s0 = {}, s1 = {};
#pragma unroll
    for (int s = 0; s < 2; ++s) {
      const uint4 b00 = *(const uint4*)(&lds_k[wave][buf][l16][s * 32 + half * 16]);
      const uint4 b01 = *(const uint4*)(&lds_k[wave][buf][l16][s * 32 + half * 16 + 8]);
      v8u b0 = {b00.x, b00.y, b00.z, b00.w, b01.x, b01.y, b01.z, b01.w};
      s0 = __builtin_amdgcn_wmma_f32_16x16x32_bf16(false, qa[s], false,
             __builtin_bit_cast(v16bf, b0), (short)0, s0, false, false);
      const uint4 b10 = *(const uint4*)(&lds_k[wave][buf][16 + l16][s * 32 + half * 16]);
      const uint4 b11 = *(const uint4*)(&lds_k[wave][buf][16 + l16][s * 32 + half * 16 + 8]);
      v8u b1 = {b10.x, b10.y, b10.z, b10.w, b11.x, b11.y, b11.z, b11.w};
      s1 = __builtin_amdgcn_wmma_f32_16x16x32_bf16(false, qa[s], false,
             __builtin_bit_cast(v16bf, b1), (short)0, s1, false, false);
    }

    // ---- online softmax; row of accumulator r is M = r + half*8 ----
#pragma unroll
    for (int r = 0; r < 8; ++r) {
      float v = fmaxf(s0[r], s1[r]);
      v = fmaxf(v, __shfl_xor(v, 1, 32));
      v = fmaxf(v, __shfl_xor(v, 2, 32));
      v = fmaxf(v, __shfl_xor(v, 4, 32));
      v = fmaxf(v, __shfl_xor(v, 8, 32));
      const float mnew  = fmaxf(m[r], v);
      const float alpha = __expf(m[r] - mnew);
      m[r] = mnew;
      const float p0 = __expf(s0[r] - mnew);
      const float p1 = __expf(s1[r] - mnew);
      float ps = p0 + p1;
      ps += __shfl_xor(ps, 1, 32);
      ps += __shfl_xor(ps, 2, 32);
      ps += __shfl_xor(ps, 4, 32);
      ps += __shfl_xor(ps, 8, 32);
      l[r] = l[r] * alpha + ps;
      o[0][r] *= alpha; o[1][r] *= alpha; o[2][r] *= alpha; o[3][r] *= alpha;
      const int row = r + half * 8;
      const unsigned int pk = bf16_pk(p0, p1);
      lds_p[wave][row][l16]      = (unsigned short)(pk & 0xFFFFu);
      lds_p[wave][row][16 + l16] = (unsigned short)(pk >> 16);
    }

    // wave-private region: enforce same-wave store->load ordering
    asm volatile("s_wait_dscnt 0" ::: "memory");

    const uint4 pa0 = *(const uint4*)(&lds_p[wave][l16][half * 16]);
    const uint4 pa1 = *(const uint4*)(&lds_p[wave][l16][half * 16 + 8]);
    v8u pu = {pa0.x, pa0.y, pa0.z, pa0.w, pa1.x, pa1.y, pa1.z, pa1.w};
    const v16bf P = __builtin_bit_cast(v16bf, pu);

    // ---- O += P @ V ; V^T fragments from the LDS tile ----
#pragma unroll
    for (int t = 0; t < 4; ++t) {
      const uint4 b0 = *(const uint4*)(&lds_v[wave][buf][t * 16 + l16][half * 16]);
      const uint4 b1 = *(const uint4*)(&lds_v[wave][buf][t * 16 + l16][half * 16 + 8]);
      v8u bu = {b0.x, b0.y, b0.z, b0.w, b1.x, b1.y, b1.z, b1.w};
      o[t] = __builtin_amdgcn_wmma_f32_16x16x32_bf16(false, P, false,
               __builtin_bit_cast(v16bf, bu), (short)0, o[t], false, false);
    }
  }
#undef ISSUE_STEP

  // ---- normalize and store fp32 output ----
#pragma unroll
  for (int r = 0; r < 8; ++r) {
    const float inv = 1.0f / l[r];
    const int row = m0 + r + half * 8;
    out[(size_t)row * D_H +  0 + l16] = o[0][r] * inv;
    out[(size_t)row * D_H + 16 + l16] = o[1][r] * inv;
    out[(size_t)row * D_H + 32 + l16] = o[2][r] * inv;
    out[(size_t)row * D_H + 48 + l16] = o[3][r] * inv;
  }
}

extern "C" void kernel_launch(void* const* d_in, const int* in_sizes, int n_in,
                              void* d_out, int out_size, void* d_ws, size_t ws_size,
                              hipStream_t stream) {
  (void)in_sizes; (void)n_in; (void)out_size; (void)ws_size;
  const float* x  = (const float*)d_in[0];
  const float* Wq = (const float*)d_in[1];
  const float* bq = (const float*)d_in[2];
  const float* Wk = (const float*)d_in[3];
  const float* bk = (const float*)d_in[4];
  const float* Wv = (const float*)d_in[5];
  const float* bv = (const float*)d_in[6];
  float* out = (float*)d_out;

  unsigned short* xb  = (unsigned short*)d_ws;
  unsigned short* wtq = xb  + (size_t)N_TOK * D_IN;
  unsigned short* wtk = wtq + (size_t)D_H * D_IN;
  unsigned short* wtv = wtk + (size_t)D_H * D_IN;
  unsigned short* qo  = wtv + (size_t)D_H * D_IN;
  unsigned short* ko  = qo  + (size_t)N_TOK * D_H;
  unsigned short* vT  = ko  + (size_t)N_TOK * D_H;

  cvt_x_kernel<<<(N_TOK * D_IN / 4) / 256, 256, 0, stream>>>(x, xb);
  cvt_w_kernel<<<768, 256, 0, stream>>>(Wq, Wk, Wv, wtq, wtk, wtv);
  qkv_kernel<<<N_TOK / 16, 128, 0, stream>>>(xb, wtq, wtk, wtv, bq, bk, bv,
                                             qo, ko, vT);
  attn_kernel<<<N_TOK / 32, 64, 0, stream>>>(qo, ko, vT, out);
}
